// LSTMText_17076789969071
// MI455X (gfx1250) — matrix operand order repaired and checked by reading, not compiled, not run on hardware
//
#include <hip/hip_runtime.h>

// ---------------- CDNA5 WMMA types ----------------
typedef __attribute__((ext_vector_type(16))) __bf16 v16bf;
typedef __attribute__((ext_vector_type(8)))  __bf16 v8bf;
typedef __attribute__((ext_vector_type(8)))  float  v8f;

#define D_DICT 50000
#define BATCH  256
#define SEQ    512
#define DH     256
#define G4     1024   // 4*DH

__device__ __forceinline__ float fsigmoid(float x) {
    return 1.0f / (1.0f + __expf(-x));
}
__device__ __forceinline__ float ftanh_fast(float x) {
    x = fminf(fmaxf(x, -10.0f), 10.0f);
    float e = __expf(2.0f * x);
    return (e - 1.0f) / (e + 1.0f);
}
__device__ __forceinline__ v8f v8f_zero() {
    v8f z;
#pragma unroll
    for (int i = 0; i < 8; ++i) z[i] = 0.0f;
    return z;
}

// Load A-fragment (16x32 bf16, M x K) per ISA layout:
// lane = half*16+l16 ; row m = l16 ; VGPR i: k = (i>>2)*16 + half*8 + (i&3)*2 (pairs)
__device__ __forceinline__ v16bf load_a_frag(const __bf16* base, int lda,
                                             int l16, int half, int k0) {
    const __bf16* ap = base + (size_t)l16 * lda + k0 + half * 8;
    v16bf a;
#pragma unroll
    for (int i = 0; i < 4; ++i) {
        a[2*i]     = ap[2*i];
        a[2*i + 1] = ap[2*i + 1];
        a[8 + 2*i] = ap[16 + 2*i];
        a[9 + 2*i] = ap[17 + 2*i];
    }
    return a;
}

// Load B-fragment (32x16 bf16, K x N) where B[k][n] = M[n0+n][k0+k] (M row-major, ldm elems):
// lane column n = l16 ; element s: k = half*8 + s (s<8), k = 16 + half*8 + (s-8) (s>=8).
__device__ __forceinline__ v16bf load_b_frag(const __bf16* mrow_base, int ldm,
                                             int l16, int half, int k0) {
    const __bf16* bp = mrow_base + (size_t)l16 * ldm + k0 + half * 8;
    v16bf b;
#pragma unroll
    for (int i = 0; i < 16; ++i) b[i] = bp[i + (i & 8)];
    return b;
}

// ---------------- fp32 -> bf16 weight conversion ----------------
__global__ __launch_bounds__(256) void cvt_bf16_kernel(const float* __restrict__ src,
                                                       __bf16* __restrict__ dst, int n) {
    int i = blockIdx.x * 256 + threadIdx.x;
    if (i < n) dst[i] = (__bf16)src[i];
}

// ---------------- Kernel 1: gather + input projection ----------------
// Computes xg = Emb[words] @ W^T + b and stores it in WMMA C/D *fragment layout*:
//   xg_frag[(s*16 + bt)*64 + gt][lane][v]  (v8bf per lane per 16x16 tile)
// so the scan kernel can reload accumulators with a single b128 per gate.
// Block: 256 threads (8 waves). Grid: 8192 blocks = (s, b-tile). Wave owns 8 column tiles.
__global__ __launch_bounds__(256) void xg_kernel(const int*   __restrict__ words,
                                                 const float* __restrict__ Emb,
                                                 const __bf16* __restrict__ Wb,
                                                 const float* __restrict__ bias,
                                                 __bf16* __restrict__ xg) {
    const int rt = blockIdx.x;          // rt = s*16 + bt
    const int s  = rt >> 4;
    const int b0 = (rt & 15) << 4;

    __shared__ __bf16 Ash[16 * 264];    // 16 rows x 256 K (stride 264: no bank conflicts)

    const int t = threadIdx.x;
    {   // stage: gather 16 embedding rows, convert fp32 -> bf16
        const int row = t >> 4;
        const int c0  = (t & 15) << 4;
        const int word = words[(size_t)(b0 + row) * SEQ + s];
        const float* src = Emb + (size_t)word * DH + c0;
#pragma unroll
        for (int i = 0; i < 16; ++i) Ash[row * 264 + c0 + i] = (__bf16)src[i];
    }
    __syncthreads();

    const int wave = t >> 5, lane = t & 31, half = lane >> 4, l16 = lane & 15;

    v8f acc[8];
#pragma unroll
    for (int j = 0; j < 8; ++j) acc[j] = v8f_zero();

#pragma unroll
    for (int k0 = 0; k0 < DH; k0 += 32) {
        v16bf a = load_a_frag(Ash, 264, l16, half, k0);
#pragma unroll
        for (int j = 0; j < 8; ++j) {
            const int g0 = (wave + 8 * j) << 4;
            v16bf b = load_b_frag(Wb + (size_t)g0 * DH, DH, l16, half, k0);
            acc[j] = __builtin_amdgcn_wmma_f32_16x16x32_bf16(
                false, a, false, b, (short)0, acc[j], false, false);
        }
    }

    // add bias, store tile fragments contiguously (16 B per lane per tile)
#pragma unroll
    for (int j = 0; j < 8; ++j) {
        const int gt = wave + 8 * j;                   // column-tile index 0..63
        const float bv = bias[(gt << 4) + l16];
        v8bf o;
#pragma unroll
        for (int v = 0; v < 8; ++v) o[v] = (__bf16)(acc[j][v] + bv);
        *(v8bf*)(xg + ((((size_t)rt * 64 + gt) * 32 + lane) << 3)) = o;
    }
}

// ---------------- Kernel 2: persistent LSTM scan ----------------
// Grid: (16 batch-chunks, 2 directions). Block: 512 threads = 16 waves.
// Each WG owns a 16-row batch slice; h double-buffered bf16 in LDS; c in fp32 registers.
// Wave w owns hidden-column slice cs=w (16 cols) across all 4 gates.
__global__ __launch_bounds__(512, 1) void lstm_kernel(const __bf16* __restrict__ xg,
                                                      const __bf16* __restrict__ Ub,
                                                      float* __restrict__ hout) {
    __shared__ __bf16 hsh[2][16 * 264];   // double-buffered h slice, padded stride

    const int chunk = blockIdx.x;           // 0..15
    const int dir   = blockIdx.y;           // 0 fwd, 1 rev

    const int t = threadIdx.x;
    const int wave = t >> 5, lane = t & 31, half = lane >> 4, l16 = lane & 15;
    const int cs = wave;                    // hidden-column tile this wave owns

    for (int i = t; i < 16 * 264; i += 512) hsh[0][i] = (__bf16)0.0f;
    __syncthreads();

    v8f creg = v8f_zero();                  // fp32 cell state, C/D layout
    int cur = 0;

#pragma unroll 1
    for (int step = 0; step < SEQ; ++step) {
        const int s = dir ? (SEQ - 1 - step) : step;
        // this chunk's xg slice for step s: 64 tiles * 32 lanes * 8 = 16384 bf16 = 32 KB
        const __bf16* xs = xg + ((size_t)(s * 16 + chunk) << 14);

        // prefetch next step's xg slice (32 KB -> 256 x 128B lines)
        if (step + 1 < SEQ) {
            const int sn = dir ? (SEQ - 2 - step) : (step + 1);
            const char* pf = (const char*)(xg + ((size_t)(sn * 16 + chunk) << 14));
            if (t < 256) __builtin_prefetch(pf + (t << 7), 0, 0);
        }

        // init accumulators from precomputed input gates: one b128 per gate
        v8f acc[4];
#pragma unroll
        for (int g = 0; g < 4; ++g) {
            const int gt = g * 16 + cs;
            v8bf xv = *(const v8bf*)(xs + (((size_t)gt * 32 + lane) << 3));
#pragma unroll
            for (int v = 0; v < 8; ++v) acc[g][v] = (float)xv[v];
        }

        // gates += h @ U^T   (K = 256 in 8 WMMA steps; A shared across the 4 gates)
#pragma unroll
        for (int k0 = 0; k0 < DH; k0 += 32) {
            v16bf a = load_a_frag(hsh[cur], 264, l16, half, k0);
#pragma unroll
            for (int g = 0; g < 4; ++g) {
                v16bf b = load_b_frag(Ub + (size_t)(g * DH + cs * 16) * DH, DH,
                                      l16, half, k0);
                acc[g] = __builtin_amdgcn_wmma_f32_16x16x32_bf16(
                    false, a, false, b, (short)0, acc[g], false, false);
            }
        }

        // activations + state update; write new h into the other LDS buffer
#pragma unroll
        for (int v = 0; v < 8; ++v) {
            const float gi = fsigmoid(acc[0][v]);
            const float gf = fsigmoid(acc[1][v]);
            const float go = fsigmoid(acc[2][v]);
            const float gg = ftanh_fast(acc[3][v]);
            const float cn = fmaf(gf, creg[v], gi * gg);
            creg[v] = cn;
            const int m = v + half * 8;
            hsh[cur ^ 1][m * 264 + cs * 16 + l16] = (__bf16)(go * ftanh_fast(cn));
        }
        __syncthreads();
        cur ^= 1;
    }

    // final hidden state -> fp32 workspace [dir][b][j]
    float* ho = hout + ((size_t)dir * BATCH + (chunk << 4)) * DH;
#pragma unroll
    for (int v = 0; v < 8; ++v) {
        const int m = v + half * 8;
        ho[(size_t)m * DH + cs * 16 + l16] = (float)hsh[cur][m * 264 + cs * 16 + l16];
    }
}

// ---------------- Kernel 3: output head ----------------
// pred[b][o] = out_b[o] + sum_j h_l[b][j]*oW[o][j] + sum_j h_r[b][j]*oW[o][256+j]
__global__ __launch_bounds__(256) void out_kernel(const float* __restrict__ hout,
                                                  const float* __restrict__ oW,
                                                  const float* __restrict__ ob,
                                                  float* __restrict__ pred) {
    const int b = threadIdx.x;
    float a0 = ob[0], a1 = ob[1];
    const float* hl = hout + (size_t)b * DH;
    const float* hr = hout + (size_t)BATCH * DH + (size_t)b * DH;
    for (int j = 0; j < DH; ++j) {
        const float x = hl[j];
        a0 = fmaf(x, oW[j], a0);
        a1 = fmaf(x, oW[2 * DH + j], a1);
    }
    for (int j = 0; j < DH; ++j) {
        const float x = hr[j];
        a0 = fmaf(x, oW[DH + j], a0);
        a1 = fmaf(x, oW[3 * DH + j], a1);
    }
    pred[b * 2 + 0] = a0;
    pred[b * 2 + 1] = a1;
}

extern "C" void kernel_launch(void* const* d_in, const int* in_sizes, int n_in,
                              void* d_out, int out_size, void* d_ws, size_t ws_size,
                              hipStream_t stream) {
    (void)in_sizes; (void)n_in; (void)out_size; (void)ws_size;
    const int*   words = (const int*)d_in[0];
    const float* Emb   = (const float*)d_in[1];
    const float* W     = (const float*)d_in[2];
    const float* U     = (const float*)d_in[3];
    const float* bias  = (const float*)d_in[4];
    const float* oW    = (const float*)d_in[5];
    const float* ob    = (const float*)d_in[6];

    // workspace layout (bytes)
    char* ws = (char*)d_ws;
    __bf16* xg   = (__bf16*)ws;                               // SEQ*BATCH*G4*2 = 256 MiB
    __bf16* Wb   = (__bf16*)(ws + (size_t)268435456);         // 512 KiB
    __bf16* Ub   = (__bf16*)(ws + (size_t)268959744);         // 512 KiB
    float*  hout = (float*) (ws + (size_t)269484032);         // 2*256*256*4 = 512 KiB

    cvt_bf16_kernel<<<dim3(1024), dim3(256), 0, stream>>>(W, Wb, G4 * DH);
    cvt_bf16_kernel<<<dim3(1024), dim3(256), 0, stream>>>(U, Ub, G4 * DH);

    xg_kernel<<<dim3(SEQ * (BATCH / 16)), dim3(256), 0, stream>>>(words, Emb, Wb, bias, xg);

    lstm_kernel<<<dim3(16, 2), dim3(512), 0, stream>>>(xg, Ub, hout);

    out_kernel<<<dim3(1), dim3(256), 0, stream>>>(hout, oW, ob, (float*)d_out);
}
